// XerxesSparseMoeBlock_49400713839219
// MI455X (gfx1250) — matrix operand order repaired and checked by reading, not compiled, not run on hardware
//
#include <hip/hip_runtime.h>
#include <hip/hip_bf16.h>
#include <math.h>
#include <cstdint>

// ---------------------------------------------------------------------------
// Problem constants (from reference): B=2, S=2048, H=2048, I=4096, E=8, TOPK=2
// ---------------------------------------------------------------------------
constexpr int TOK  = 4096;   // B*S tokens
constexpr int HDIM = 2048;
constexpr int IDIM = 4096;
constexpr int NEXP = 8;

constexpr int MT = 64;       // token tile (64 -> halves weight re-reads vs 32)
constexpr int NT = 128;      // output-feature tile
constexpr int KT = 32;       // K step (matches 16x16x32 bf16 WMMA)
constexpr int LDK = 40;      // padded K row (40 bf16 = 80B = 5*16B -> aligned b128)

typedef __attribute__((ext_vector_type(16))) __bf16 v16bf;
typedef __attribute__((ext_vector_type(8)))  __bf16 v8bf;
typedef __attribute__((ext_vector_type(8)))  float  v8f;

// ---------------------------------------------------------------------------
// CDNA5-specific helpers
// ---------------------------------------------------------------------------
// Packed f32->bf16 (RNE) : one VALU op for two elements.
__device__ inline uint32_t pk_bf16(float a, float b) {
    uint32_t r;
    asm volatile("v_cvt_pk_bf16_f32 %0, %1, %2" : "=v"(r) : "v"(a), "v"(b));
    return r;
}
__device__ inline void store_pk_bf16(__bf16* d0, __bf16* d1, float a, float b) {
    union { uint32_t u; __bf16 bb[2]; } c;
    c.u = pk_bf16(a, b);
    *d0 = c.bb[0];
    *d1 = c.bb[1];
}
// Async global->LDS 16B copy (ASYNCcnt-tracked, bypasses VGPRs).
__device__ inline void async_copy_b128(uint32_t lds_off, const void* gptr) {
    asm volatile("global_load_async_to_lds_b128 %0, %1, off"
                 :: "v"(lds_off), "v"(gptr) : "memory");
}
__device__ inline void wait_async0() {
    asm volatile("s_wait_asynccnt 0x0" ::: "memory");
}
__device__ inline uint32_t lds_offset_of(const void* p) {
    return (uint32_t)(uintptr_t)p;   // low 32 bits of generic ptr = LDS address
}

__device__ inline float gelu_tanh(float v) {
    const float k0 = 0.7978845608028654f;   // sqrt(2/pi)
    const float k1 = 0.044715f;
    return 0.5f * v * (1.0f + tanhf(k0 * (v + k1 * v * v * v)));
}

// Build a 16-element bf16 fragment from two aligned 8-element LDS chunks.
__device__ inline v16bf frag16(const __bf16* p_lo, const __bf16* p_hi) {
    v8bf lo = *(const v8bf*)p_lo;
    v8bf hi = *(const v8bf*)p_hi;
    v16bf r;
#pragma unroll
    for (int i = 0; i < 8; ++i) { r[i] = lo[i]; r[i + 8] = hi[i]; }
    return r;
}

// ---------------------------------------------------------------------------
// Kernel: convert x (f32) -> bf16 workspace (packed, 2 elems per thread-iter)
// ---------------------------------------------------------------------------
__global__ void cvt_x_kernel(const float* __restrict__ x,
                             uint32_t* __restrict__ xb2, int npairs) {
    for (int i = blockIdx.x * blockDim.x + threadIdx.x; i < npairs;
         i += gridDim.x * blockDim.x) {
        float2 v = ((const float2*)x)[i];
        xb2[i] = pk_bf16(v.x, v.y);
    }
}

// ---------------------------------------------------------------------------
// Kernel: router logits  (T x H) @ (H x E) -> logits tail of d_out
// ---------------------------------------------------------------------------
__global__ void router_logits_kernel(const float* __restrict__ x,
                                     const float* __restrict__ gw,
                                     float* __restrict__ logits) {
    int gid = blockIdx.x * blockDim.x + threadIdx.x;
    if (gid >= TOK * NEXP) return;
    int t = gid >> 3, e = gid & 7;
    const float* xr = x + (size_t)t * HDIM;
    float acc = 0.f;
    for (int h = 0; h < HDIM; ++h) acc = fmaf(xr[h], gw[h * NEXP + e], acc);
    logits[gid] = acc;
}

// ---------------------------------------------------------------------------
// Kernel: top-2 + softmax -> per-token coefficient vector coeff[t][e]
// ---------------------------------------------------------------------------
__global__ void router_coeff_kernel(const float* __restrict__ logits,
                                    float* __restrict__ coeff) {
    int t = blockIdx.x * blockDim.x + threadIdx.x;
    if (t >= TOK) return;
    float l[NEXP];
#pragma unroll
    for (int e = 0; e < NEXP; ++e) l[e] = logits[t * NEXP + e];
    int i0 = 0;
#pragma unroll
    for (int e = 1; e < NEXP; ++e) if (l[e] > l[i0]) i0 = e;   // ties: lowest idx
    int i1 = (i0 == 0) ? 1 : 0;
#pragma unroll
    for (int e = 0; e < NEXP; ++e)
        if (e != i0 && l[e] > l[i1]) i1 = e;
    float e1 = __expf(l[i1] - l[i0]);
    float inv = 1.0f / (1.0f + e1);
#pragma unroll
    for (int e = 0; e < NEXP; ++e)
        coeff[t * NEXP + e] = (e == i0) ? inv : (e == i1) ? e1 * inv : 0.0f;
}

// ---------------------------------------------------------------------------
// Kernel: fused gate+up GEMM for one expert.
//   act[t,i] = bf16( gelu(x@Wg) * (x@Wu) )
// Block = 64(tok) x 128(I); 8 wave32 waves (2x4), each owns a 32x32 subtile
// for both gate and up => 8 WMMAs per K step per wave.
// ---------------------------------------------------------------------------
__global__ void __launch_bounds__(256)
moe_gate_up_kernel(const unsigned short* __restrict__ xb,    // [TOK][HDIM] bf16
                   const float* __restrict__ wg,             // [HDIM][IDIM]
                   const float* __restrict__ wu,             // [HDIM][IDIM]
                   unsigned short* __restrict__ act) {       // [TOK][IDIM] bf16
    __shared__ __align__(16) __bf16 As[MT][LDK];   //  5.1 KB
    __shared__ __align__(16) __bf16 Bg[NT][LDK];   // 10.2 KB
    __shared__ __align__(16) __bf16 Bu[NT][LDK];   // 10.2 KB

    const int t0   = blockIdx.y * MT;
    const int n0   = blockIdx.x * NT;
    const int tid  = threadIdx.x;
    const int lane = tid & 31;
    const int wave = tid >> 5;
    const int wmB  = (wave >> 2) << 5;   // 0 / 32  (M half)
    const int wn   = (wave & 3) << 5;    // 0 / 32 / 64 / 96

    v8f accg[2][2] = {};   // [m-sub][n-sub] gate
    v8f accu[2][2] = {};   // [m-sub][n-sub] up

    // A tile: 64x32 bf16 = 4 KB -> one async b128 per thread.
    const int aidx = tid << 3;           // 8 bf16 per thread
    const int ar = aidx >> 5, ac = aidx & 31;
    const uint32_t a_lds = lds_offset_of(&As[ar][ac]);
    const unsigned short* a_g = xb + (size_t)(t0 + ar) * HDIM + ac;

    for (int k0 = 0; k0 < HDIM; k0 += KT) {
        async_copy_b128(a_lds, a_g + k0);

        // B tiles: 32(K)x128(N) f32 -> transposed bf16 [N][K], packed converts
#pragma unroll
        for (int j = 0; j < 4; ++j) {
            int linear = tid + 256 * j;          // 0..1023 float4 chunks
            int k  = linear >> 5;                // 0..31
            int nq = (linear & 31) << 2;         // 0,4,...,124
            size_t goff = (size_t)(k0 + k) * IDIM + n0 + nq;
            float4 g = *(const float4*)(wg + goff);
            float4 u = *(const float4*)(wu + goff);
            store_pk_bf16(&Bg[nq + 0][k], &Bg[nq + 1][k], g.x, g.y);
            store_pk_bf16(&Bg[nq + 2][k], &Bg[nq + 3][k], g.z, g.w);
            store_pk_bf16(&Bu[nq + 0][k], &Bu[nq + 1][k], u.x, u.y);
            store_pk_bf16(&Bu[nq + 2][k], &Bu[nq + 3][k], u.z, u.w);
            if (k0 + KT < HDIM)                  // warm L2 for next tile
                __builtin_prefetch(wg + goff + (size_t)KT * IDIM, 0, 1);
        }
        wait_async0();
        __syncthreads();

        // fragments: contiguous k = 16*(lane/16) + s, same order for A and B
        const int arow = wmB + (lane & 15);
        const int aklo = (lane >> 4) << 4;                 // 0 / 16
        v16bf a0 = frag16(&As[arow][aklo],      &As[arow][aklo + 8]);
        v16bf a1 = frag16(&As[arow + 16][aklo], &As[arow + 16][aklo + 8]);

        const int bcol = lane & 15;
        const int bk   = (lane >> 4) << 4;                 // 0 / 16
        v16bf b0 = frag16(&Bg[wn + bcol][bk],      &Bg[wn + bcol][bk + 8]);
        v16bf b1 = frag16(&Bg[wn + 16 + bcol][bk], &Bg[wn + 16 + bcol][bk + 8]);
        accg[0][0] = __builtin_amdgcn_wmma_f32_16x16x32_bf16(false, a0, false, b0,
                                                             (short)0, accg[0][0], false, false);
        accg[0][1] = __builtin_amdgcn_wmma_f32_16x16x32_bf16(false, a0, false, b1,
                                                             (short)0, accg[0][1], false, false);
        accg[1][0] = __builtin_amdgcn_wmma_f32_16x16x32_bf16(false, a1, false, b0,
                                                             (short)0, accg[1][0], false, false);
        accg[1][1] = __builtin_amdgcn_wmma_f32_16x16x32_bf16(false, a1, false, b1,
                                                             (short)0, accg[1][1], false, false);

        b0 = frag16(&Bu[wn + bcol][bk],      &Bu[wn + bcol][bk + 8]);
        b1 = frag16(&Bu[wn + 16 + bcol][bk], &Bu[wn + 16 + bcol][bk + 8]);
        accu[0][0] = __builtin_amdgcn_wmma_f32_16x16x32_bf16(false, a0, false, b0,
                                                             (short)0, accu[0][0], false, false);
        accu[0][1] = __builtin_amdgcn_wmma_f32_16x16x32_bf16(false, a0, false, b1,
                                                             (short)0, accu[0][1], false, false);
        accu[1][0] = __builtin_amdgcn_wmma_f32_16x16x32_bf16(false, a1, false, b0,
                                                             (short)0, accu[1][0], false, false);
        accu[1][1] = __builtin_amdgcn_wmma_f32_16x16x32_bf16(false, a1, false, b1,
                                                             (short)0, accu[1][1], false, false);
        __syncthreads();
    }

    // epilogue: gelu(g)*u, packed bf16 convert + b16 stores.
    // C layout: VGPR i -> M = i + 8*(lane>=16), N = lane%16.
    const int mofs = (lane >> 4) << 3;
    const int ncol = lane & 15;
#pragma unroll
    for (int ms = 0; ms < 2; ++ms) {
#pragma unroll
        for (int i = 0; i < 8; ++i) {
            int m = t0 + wmB + (ms << 4) + mofs + i;
            int n = n0 + wn + ncol;
            float v0 = gelu_tanh(accg[ms][0][i]) * accu[ms][0][i];
            float v1 = gelu_tanh(accg[ms][1][i]) * accu[ms][1][i];
            union { uint32_t u; unsigned short s[2]; } p;
            p.u = pk_bf16(v0, v1);
            act[(size_t)m * IDIM + n]      = p.s[0];
            act[(size_t)m * IDIM + n + 16] = p.s[1];
        }
    }
}

// ---------------------------------------------------------------------------
// Kernel: down GEMM for one expert, fused routing-weight accumulate into out.
//   out[t,h] = (first ? 0 : out[t,h]) + coeff[t,e] * (act @ Wd)[t,h]
// ---------------------------------------------------------------------------
__global__ void __launch_bounds__(256)
moe_down_kernel(const unsigned short* __restrict__ act,   // [TOK][IDIM] bf16
                const float* __restrict__ wd,             // [IDIM][HDIM]
                const float* __restrict__ coeff,          // [TOK][NEXP]
                float* __restrict__ out,                  // [TOK][HDIM]
                int e, int is_first) {
    __shared__ __align__(16) __bf16 As[MT][LDK];
    __shared__ __align__(16) __bf16 Bs[NT][LDK];
    __shared__ float cf[MT];

    const int t0   = blockIdx.y * MT;
    const int n0   = blockIdx.x * NT;
    const int tid  = threadIdx.x;
    const int lane = tid & 31;
    const int wave = tid >> 5;
    const int wmB  = (wave >> 2) << 5;
    const int wn   = (wave & 3) << 5;

    if (tid < MT) cf[tid] = coeff[(size_t)(t0 + tid) * NEXP + e];

    v8f c[2][2] = {};

    const int aidx = tid << 3;
    const int ar = aidx >> 5, ac = aidx & 31;
    const uint32_t a_lds = lds_offset_of(&As[ar][ac]);
    const unsigned short* a_g = act + (size_t)(t0 + ar) * IDIM + ac;

    for (int k0 = 0; k0 < IDIM; k0 += KT) {
        async_copy_b128(a_lds, a_g + k0);
#pragma unroll
        for (int j = 0; j < 4; ++j) {
            int linear = tid + 256 * j;
            int k  = linear >> 5;
            int nq = (linear & 31) << 2;
            size_t goff = (size_t)(k0 + k) * HDIM + n0 + nq;
            float4 d = *(const float4*)(wd + goff);
            store_pk_bf16(&Bs[nq + 0][k], &Bs[nq + 1][k], d.x, d.y);
            store_pk_bf16(&Bs[nq + 2][k], &Bs[nq + 3][k], d.z, d.w);
            if (k0 + KT < IDIM)
                __builtin_prefetch(wd + goff + (size_t)KT * HDIM, 0, 1);
        }
        wait_async0();
        __syncthreads();

        const int arow = wmB + (lane & 15);
        const int aklo = (lane >> 4) << 4;
        v16bf a0 = frag16(&As[arow][aklo],      &As[arow][aklo + 8]);
        v16bf a1 = frag16(&As[arow + 16][aklo], &As[arow + 16][aklo + 8]);

        const int bcol = lane & 15;
        const int bk   = (lane >> 4) << 4;
        v16bf b0 = frag16(&Bs[wn + bcol][bk],      &Bs[wn + bcol][bk + 8]);
        v16bf b1 = frag16(&Bs[wn + 16 + bcol][bk], &Bs[wn + 16 + bcol][bk + 8]);

        c[0][0] = __builtin_amdgcn_wmma_f32_16x16x32_bf16(false, a0, false, b0,
                                                          (short)0, c[0][0], false, false);
        c[0][1] = __builtin_amdgcn_wmma_f32_16x16x32_bf16(false, a0, false, b1,
                                                          (short)0, c[0][1], false, false);
        c[1][0] = __builtin_amdgcn_wmma_f32_16x16x32_bf16(false, a1, false, b0,
                                                          (short)0, c[1][0], false, false);
        c[1][1] = __builtin_amdgcn_wmma_f32_16x16x32_bf16(false, a1, false, b1,
                                                          (short)0, c[1][1], false, false);
        __syncthreads();
    }

    const int mofs = (lane >> 4) << 3;
    const int ncol = lane & 15;
#pragma unroll
    for (int ms = 0; ms < 2; ++ms) {
#pragma unroll
        for (int i = 0; i < 8; ++i) {
            int mrel = wmB + (ms << 4) + mofs + i;
            float w  = cf[mrel];
            size_t o = (size_t)(t0 + mrel) * HDIM + n0 + wn + ncol;
            float p0 = is_first ? 0.0f : out[o];
            float p1 = is_first ? 0.0f : out[o + 16];
            out[o]      = p0 + w * c[ms][0][i];
            out[o + 16] = p1 + w * c[ms][1][i];
        }
    }
}

// ---------------------------------------------------------------------------
// Host launcher
// ---------------------------------------------------------------------------
extern "C" void kernel_launch(void* const* d_in, const int* in_sizes, int n_in,
                              void* d_out, int out_size, void* d_ws, size_t ws_size,
                              hipStream_t stream) {
    const float* x         = (const float*)d_in[0];  // [B,S,H]
    const float* gate_w    = (const float*)d_in[1];  // [H,E]
    const float* gate_proj = (const float*)d_in[2];  // [E,H,I]
    const float* up_proj   = (const float*)d_in[3];  // [E,H,I]
    const float* down_proj = (const float*)d_in[4];  // [E,I,H]

    float* out    = (float*)d_out;                       // final [T,H] ...
    float* logits = out + (size_t)TOK * HDIM;            // ... then logits [T,E]

    char* ws = (char*)d_ws;
    unsigned short* xb  = (unsigned short*)ws;                               // 16 MB
    unsigned short* act = (unsigned short*)(ws + (size_t)TOK * HDIM * 2);    // 32 MB
    float* coeff = (float*)(ws + (size_t)TOK * HDIM * 2 + (size_t)TOK * IDIM * 2);

    cvt_x_kernel<<<2048, 256, 0, stream>>>(x, (uint32_t*)xb, TOK * HDIM / 2);
    router_logits_kernel<<<(TOK * NEXP + 255) / 256, 256, 0, stream>>>(x, gate_w, logits);
    router_coeff_kernel<<<(TOK + 255) / 256, 256, 0, stream>>>(logits, coeff);

    for (int e = 0; e < NEXP; ++e) {
        const float* wg = gate_proj + (size_t)e * HDIM * IDIM;
        const float* wu = up_proj   + (size_t)e * HDIM * IDIM;
        const float* wd = down_proj + (size_t)e * IDIM * HDIM;
        moe_gate_up_kernel<<<dim3(IDIM / NT, TOK / MT), 256, 0, stream>>>(xb, wg, wu, act);
        moe_down_kernel<<<dim3(HDIM / NT, TOK / MT), 256, 0, stream>>>(act, wd, coeff, out,
                                                                       e, e == 0 ? 1 : 0);
    }
}
